// WanSelfAttention_28810640621737
// MI455X (gfx1250) — matrix operand order repaired and checked by reading, not compiled, not run on hardware
//
#include <hip/hip_runtime.h>
#include <hip/hip_bf16.h>
#include <cstdint>

// ---------------------------------------------------------------------------
// WanSelfAttention for MI455X (gfx1250): bf16 WMMA everywhere.
// GEMM tiles staged by the Tensor Data Mover (tensor_load_to_lds, TENSORcnt)
// with hardware LDS padding; attention K/V staged by async global->LDS
// (GLOBAL_LOAD_ASYNC_TO_LDS_B128, ASYNCcnt).
// B=1, S=4096, DIM=1536, HEADS=12, HD=128, rope C=64 (split 22/21/21).
// ---------------------------------------------------------------------------

#define S_LEN   4096
#define DIM_    1536
#define HEADS_  12
#define HD_     128

typedef __attribute__((ext_vector_type(16))) __bf16    bf16x16;
typedef __attribute__((ext_vector_type(8)))  float     f32x8;
typedef __attribute__((ext_vector_type(4)))  unsigned  u32x4;
typedef __attribute__((ext_vector_type(8)))  unsigned  u32x8;

union Frag { bf16x16 v; uint4 q[2]; };

__device__ __forceinline__ __bf16 f2bf(float f) {
    union { float f; unsigned u; } a; a.f = f;
    unsigned r = a.u + 0x7FFFu + ((a.u >> 16) & 1u);   // round-to-nearest-even
    union { unsigned short s; __bf16 b; } o; o.s = (unsigned short)(r >> 16);
    return o.b;
}

__device__ __forceinline__ f32x8 zero8() {
    f32x8 z;
#pragma unroll
    for (int i = 0; i < 8; ++i) z[i] = 0.f;
    return z;
}

// ---- CDNA5 async global->LDS copy (16B per lane), tracked by ASYNCcnt ----
__device__ __forceinline__ void async_copy_b128(void* lds, const void* gptr) {
    unsigned l = (unsigned)(uintptr_t)lds;   // low 32 bits = LDS byte address
    asm volatile("global_load_async_to_lds_b128 %0, %1, off"
                 :: "v"(l), "v"(gptr) : "memory");
}
__device__ __forceinline__ void wait_async0() {
    asm volatile("s_wait_asynccnt 0x0" ::: "memory");
}

// ---- CDNA5 Tensor Data Mover: 2D tile global->LDS with LDS padding --------
// D# per CDNA5 ISA. 2-D tensor => groups 2/3 NULL.
// data_size = 2 bytes (bf16). pad inserted every (2<<pad_int_code) DWORDs,
// amount (pad_amt_code+1) DWORDs.
__device__ __forceinline__ void tdm_load_tile_2d(
    unsigned lds_addr, const void* gaddr,
    unsigned tensor_d0, unsigned tensor_d1,
    unsigned tile_d0, unsigned tile_d1, unsigned stride0,
    unsigned pad_int_code, unsigned pad_amt_code)
{
    unsigned long long ga = (unsigned long long)(uintptr_t)gaddr;
    u32x4 g0;
    g0.x = 1u;                                   // count=1 valid descriptor
    g0.y = lds_addr;                             // [63:32] LDS byte address
    g0.z = (unsigned)ga;                         // global_addr[31:0]
    g0.w = (unsigned)(ga >> 32) | 0x80000000u;   // addr[56:32] | type=2<<30
    u32x8 g1;
    g1[0] = (1u << 16)                           // data_size = 1 (2 bytes)
          | (1u << 20)                           // pad_enable
          | (pad_int_code << 22) | (pad_amt_code << 25);
    g1[1] = (tensor_d0 & 0xFFFFu) << 16;         // tensor_dim0[15:0]
    g1[2] = ((tensor_d0 >> 16) & 0xFFFFu) | ((tensor_d1 & 0xFFFFu) << 16);
    g1[3] = ((tensor_d1 >> 16) & 0xFFFFu) | ((tile_d0 & 0xFFFFu) << 16);
    g1[4] = tile_d1 & 0xFFFFu;                   // tile_dim1; tile_dim2 = 0
    g1[5] = stride0;                             // tensor_dim0_stride[31:0]
    g1[6] = 0;                                   // stride0 hi, dim1_stride lo
    g1[7] = 0;
    asm volatile("tensor_load_to_lds %0, %1" :: "s"(g0), "s"(g1) : "memory");
}

// A fragment (16x32, row-per-lane, K interleave 0-7/16-23 vs 8-15/24-31)
__device__ __forceinline__ bf16x16 ldA(const __bf16* base, int stride) {
    int lane = threadIdx.x & 31;
    int r    = lane & 15;
    int klo  = (lane & 16) >> 1;          // 0 or 8
    const __bf16* p = base + r * stride + klo;
    Frag f;
    f.q[0] = *(const uint4*)(p);          // K = klo .. klo+7
    f.q[1] = *(const uint4*)(p + 16);     // K = klo+16 .. klo+23
    return f.v;
}

// B fragment (32x16) loaded from B^T stored row-major (row = column n).
// Lane n holds K = 0-7/16-23 (lanes 0-15) or K = 8-15/24-31 (lanes 16-31).
__device__ __forceinline__ bf16x16 ldBt(const __bf16* base, int stride) {
    int lane = threadIdx.x & 31;
    int n    = lane & 15;
    int ks   = (lane & 16) >> 1;          // 0 or 8
    const __bf16* p = base + n * stride + ks;
    Frag f;
    f.q[0] = *(const uint4*)(p);
    f.q[1] = *(const uint4*)(p + 16);
    return f.v;
}

// ---------------------------------------------------------------------------
__global__ void cvt_f32_bf16(const float* __restrict__ in, __bf16* __restrict__ out, int n) {
    int i = blockIdx.x * 256 + threadIdx.x;
    if (i < n) out[i] = f2bf(in[i]);
}

// ---------------------------------------------------------------------------
// C[M,N] = A[M,K] @ W[N,K]^T + bias   (A, W bf16 row-major; C fp32)
#define TM 128
#define TN 128
#define TK 32
#define LDT 40   // 32 data + 8 pad bf16 per row; TDM pad: 16 DW int, 4 DW amt

__global__ __launch_bounds__(256)
void gemm_bf16(const __bf16* __restrict__ A, const __bf16* __restrict__ W,
               const float* __restrict__ bias, float* __restrict__ C,
               int M, int N, int K)
{
    __shared__ __bf16 lds_a[TM * LDT];
    __shared__ __bf16 lds_w[TN * LDT];

    const int tid  = threadIdx.x;
    const int lane = tid & 31;
    const int wave = tid >> 5;
    const int m0   = blockIdx.x * TM;
    const int n0   = blockIdx.y * TN;

    f32x8 acc[8];
#pragma unroll
    for (int j = 0; j < 8; ++j) acc[j] = zero8();

    for (int k0 = 0; k0 < K; k0 += TK) {
        __syncthreads();
        // TDM-stage both tiles: wave 0 issues two tensor DMAs, padding the
        // 64B rows out to 80B (LDT) in LDS; everyone else waits on barrier.
        if (wave == 0) {
            tdm_load_tile_2d((unsigned)(uintptr_t)lds_a,
                             A + (size_t)m0 * K + k0,
                             (unsigned)K, (unsigned)M, TK, TM, (unsigned)K,
                             /*pad 16 DW*/3, /*amt 4 DW*/3);
            tdm_load_tile_2d((unsigned)(uintptr_t)lds_w,
                             W + (size_t)n0 * K + k0,
                             (unsigned)K, (unsigned)N, TK, TN, (unsigned)K,
                             3, 3);
            __builtin_amdgcn_s_wait_tensorcnt(0);
        }
        __syncthreads();

        bf16x16 af = ldA(lds_a + wave * 16 * LDT, LDT);
#pragma unroll
        for (int j = 0; j < 8; ++j) {
            bf16x16 wf = ldBt(lds_w + j * 16 * LDT, LDT);
            acc[j] = __builtin_amdgcn_wmma_f32_16x16x32_bf16(
                false, af, false, wf, (short)0, acc[j], false, false);
        }
    }

    const int row_half = (lane >> 4) << 3;   // 0 or 8
    const int col_l    = lane & 15;
#pragma unroll
    for (int j = 0; j < 8; ++j) {
#pragma unroll
        for (int i = 0; i < 8; ++i) {
            int row = m0 + wave * 16 + i + row_half;
            int col = n0 + j * 16 + col_l;
            C[(size_t)row * N + col] = acc[j][i] + bias[col];
        }
    }
}

// ---------------------------------------------------------------------------
// RMSNorm over full 1536 row + 3D RoPE (per head of 128), write bf16.
__global__ __launch_bounds__(256)
void norm_rope(const float* __restrict__ in, const float* __restrict__ w,
               const float* __restrict__ fcos, const float* __restrict__ fsin,
               __bf16* __restrict__ out, int do_norm, int do_rope)
{
    const int s   = blockIdx.x;
    const int tid = threadIdx.x;
    const float* row = in + (size_t)s * DIM_;

    __shared__ float red[256];
    float ss = 0.f;
    for (int i = tid; i < DIM_; i += 256) { float v = row[i]; ss += v * v; }
    red[tid] = ss;
    __syncthreads();
    for (int off = 128; off; off >>= 1) {
        if (tid < off) red[tid] += red[tid + off];
        __syncthreads();
    }
    const float rs = do_norm ? rsqrtf(red[0] / (float)DIM_ + 1e-6f) : 1.f;

    // 3D rope row selection (grid 16x16x16; video 3840 tokens + 256 cond)
    const int frow = (s < 3840) ? (s >> 8) : 1023;
    const int hrow = (s >> 4) & 15;
    const int wrow = s & 15;

    for (int p = tid; p < DIM_ / 2; p += 256) {
        int   d0 = p * 2;
        int   i  = (d0 & (HD_ - 1)) >> 1;   // pair index within head, 0..63
        float a  = row[d0], b = row[d0 + 1];
        if (do_norm) { a *= rs * w[d0]; b *= rs * w[d0 + 1]; }
        float oa = a, ob = b;
        if (do_rope) {
            int   rr = (i < 22) ? frow : (i < 43) ? hrow : wrow;
            float c  = fcos[rr * 64 + i];
            float sn = fsin[rr * 64 + i];
            oa = a * c - b * sn;
            ob = a * sn + b * c;
        }
        out[(size_t)s * DIM_ + d0]     = f2bf(oa);
        out[(size_t)s * DIM_ + d0 + 1] = f2bf(ob);
    }
}

// ---------------------------------------------------------------------------
// Flash attention: one WG per (query block of 128, head). 8 waves x 16 rows.
#define KB 64
#define LVS 136   // padded LDS row stride for K/V tiles (bf16)

__global__ __launch_bounds__(256)
void flash_attn(const __bf16* __restrict__ Q, const __bf16* __restrict__ Km,
                const __bf16* __restrict__ V, __bf16* __restrict__ O,
                const int* __restrict__ seq_lens)
{
    const int head = blockIdx.y;
    const int tid  = threadIdx.x;
    const int lane = tid & 31;
    const int wave = tid >> 5;
    const int q0   = blockIdx.x * 128 + wave * 16;
    const int slen = seq_lens[0];
    const size_t rstride = (size_t)HEADS_ * HD_;   // 1536

    const __bf16* Qh = Q + head * HD_;
    const __bf16* Kh = Km + head * HD_;
    const __bf16* Vh = V + head * HD_;

    __shared__ __bf16 lds_k[KB * LVS];
    __shared__ __bf16 lds_v[KB * LVS];
    __shared__ __bf16 lds_p[8][16 * 64];   // per-wave P transpose scratch

    // Preload Q fragments (4 K-slices of 32 over hd=128) into registers.
    bf16x16 qf[4];
    {
        int r = lane & 15, klo = (lane & 16) >> 1;
        const __bf16* qp = Qh + (size_t)(q0 + r) * rstride;
#pragma unroll
        for (int kk = 0; kk < 4; ++kk) {
            Frag f;
            f.q[0] = *(const uint4*)(qp + kk * 32 + klo);
            f.q[1] = *(const uint4*)(qp + kk * 32 + klo + 16);
            qf[kk] = f.v;
        }
    }

    f32x8 oacc[8];
#pragma unroll
    for (int j = 0; j < 8; ++j) oacc[j] = zero8();
    float mrow[8], lrow[8];
#pragma unroll
    for (int i = 0; i < 8; ++i) { mrow[i] = -1e30f; lrow[i] = 0.f; }

    const float scale = 0.08838834764831845f;   // 1/sqrt(128)
    const int   col_l = lane & 15;
    const int   row_half = (lane >> 4) << 3;

    for (int kb0 = 0; kb0 < S_LEN; kb0 += KB) {
        __syncthreads();
        // Async-stage K/V block: 64 rows x 128 bf16 each (1024 x 16B, 4/thread)
#pragma unroll
        for (int it = 0; it < 4; ++it) {
            int idx = tid + it * 256;
            int r = idx >> 4, c4 = idx & 15;
            async_copy_b128(&lds_k[r * LVS + c4 * 8],
                            Kh + (size_t)(kb0 + r) * rstride + c4 * 8);
            async_copy_b128(&lds_v[r * LVS + c4 * 8],
                            Vh + (size_t)(kb0 + r) * rstride + c4 * 8);
        }
        if (kb0 + KB < S_LEN)
            __builtin_prefetch(Kh + (size_t)(kb0 + KB + (tid >> 2)) * rstride, 0, 1);
        wait_async0();
        __syncthreads();

        // ---- scores: 4 tiles of 16 keys, K=128 in 4 WMMA steps each ----
        f32x8 sc[4];
#pragma unroll
        for (int j = 0; j < 4; ++j) {
            f32x8 a = zero8();
#pragma unroll
            for (int kk = 0; kk < 4; ++kk) {
                bf16x16 kf = ldBt(lds_k + j * 16 * LVS + kk * 32, LVS);
                a = __builtin_amdgcn_wmma_f32_16x16x32_bf16(
                    false, qf[kk], false, kf, (short)0, a, false, false);
            }
#pragma unroll
            for (int e = 0; e < 8; ++e) a[e] *= scale;
            // mask keys beyond seq_len
            if (kb0 + j * 16 + col_l >= slen)
#pragma unroll
                for (int e = 0; e < 8; ++e) a[e] = -1e30f;
            sc[j] = a;
        }

        // ---- online softmax ----
        float bmax[8];
#pragma unroll
        for (int i = 0; i < 8; ++i) bmax[i] = -1e30f;
#pragma unroll
        for (int j = 0; j < 4; ++j)
#pragma unroll
            for (int i = 0; i < 8; ++i) bmax[i] = fmaxf(bmax[i], sc[j][i]);
#pragma unroll
        for (int i = 0; i < 8; ++i) {           // reduce across 16-lane half
            float v = bmax[i];
            v = fmaxf(v, __shfl_xor(v, 1, 32));
            v = fmaxf(v, __shfl_xor(v, 2, 32));
            v = fmaxf(v, __shfl_xor(v, 4, 32));
            v = fmaxf(v, __shfl_xor(v, 8, 32));
            float nm    = fmaxf(mrow[i], v);
            float alpha = __expf(mrow[i] - nm);
            mrow[i] = nm;
            lrow[i] *= alpha;
#pragma unroll
            for (int j = 0; j < 8; ++j) oacc[j][i] *= alpha;
        }
        float rsum[8];
#pragma unroll
        for (int i = 0; i < 8; ++i) rsum[i] = 0.f;
#pragma unroll
        for (int j = 0; j < 4; ++j)
#pragma unroll
            for (int i = 0; i < 8; ++i) {
                float p = __expf(sc[j][i] - mrow[i]);
                sc[j][i] = p;
                rsum[i] += p;
            }
#pragma unroll
        for (int i = 0; i < 8; ++i) {
            float r = rsum[i];
            r += __shfl_xor(r, 1, 32);
            r += __shfl_xor(r, 2, 32);
            r += __shfl_xor(r, 4, 32);
            r += __shfl_xor(r, 8, 32);
            lrow[i] += r;
        }

        // ---- transpose P through per-wave LDS into A-fragment layout ----
        __bf16* pbuf = &lds_p[wave][0];   // 16 rows x 64 keys
#pragma unroll
        for (int j = 0; j < 4; ++j)
#pragma unroll
            for (int i = 0; i < 8; ++i)
                pbuf[(i + row_half) * 64 + j * 16 + col_l] = f2bf(sc[j][i]);

        // ---- P @ V : 2 key-slices of 32, 8 output tiles ----
#pragma unroll
        for (int t = 0; t < 2; ++t) {
            bf16x16 pf = ldA(pbuf + t * 32, 64);
#pragma unroll
            for (int j = 0; j < 8; ++j) {
                bf16x16 vf;
                int n = lane & 15, ks = (lane & 16) >> 1;
#pragma unroll
                for (int e = 0; e < 16; ++e)
                    vf[e] = lds_v[(t * 32 + ks + (e < 8 ? e : e + 8)) * LVS + j * 16 + n];
                oacc[j] = __builtin_amdgcn_wmma_f32_16x16x32_bf16(
                    false, pf, false, vf, (short)0, oacc[j], false, false);
            }
        }
    }

    // ---- epilogue: normalize by l, write bf16 [s, head*128 + d] ----
#pragma unroll
    for (int j = 0; j < 8; ++j)
#pragma unroll
        for (int i = 0; i < 8; ++i) {
            int srow = q0 + i + row_half;
            float v = oacc[j][i] / lrow[i];
            O[(size_t)srow * rstride + head * HD_ + j * 16 + col_l] = f2bf(v);
        }
}

// ---------------------------------------------------------------------------
extern "C" void kernel_launch(void* const* d_in, const int* in_sizes, int n_in,
                              void* d_out, int out_size, void* d_ws, size_t ws_size,
                              hipStream_t stream)
{
    const float* x   = (const float*)d_in[0];
    const float* Wq  = (const float*)d_in[1];
    const float* bq  = (const float*)d_in[2];
    const float* Wk  = (const float*)d_in[3];
    const float* bk  = (const float*)d_in[4];
    const float* Wv  = (const float*)d_in[5];
    const float* bv  = (const float*)d_in[6];
    const float* Wo  = (const float*)d_in[7];
    const float* bo  = (const float*)d_in[8];
    const float* nqw = (const float*)d_in[9];
    const float* nkw = (const float*)d_in[10];
    const float* fc  = (const float*)d_in[11];
    const float* fs  = (const float*)d_in[12];
    const int*   sl  = (const int*)d_in[13];

    const size_t SD   = (size_t)S_LEN * DIM_;
    const size_t DD   = (size_t)DIM_ * DIM_;
    size_t off = 0;
    auto take = [&](size_t bytes) {
        void* p = (char*)d_ws + off;
        off += (bytes + 255) & ~(size_t)255;
        return p;
    };
    __bf16* xb  = (__bf16*)take(SD * 2);
    __bf16* wqb = (__bf16*)take(DD * 2);
    __bf16* wkb = (__bf16*)take(DD * 2);
    __bf16* wvb = (__bf16*)take(DD * 2);
    __bf16* wob = (__bf16*)take(DD * 2);
    float*  tmp = (float*)take(SD * 4);     // reused fp32 GEMM output
    __bf16* qb  = (__bf16*)take(SD * 2);
    __bf16* kb  = (__bf16*)take(SD * 2);
    __bf16* vb  = (__bf16*)take(SD * 2);
    __bf16* ab  = (__bf16*)take(SD * 2);

    // 1) fp32 -> bf16 conversions
    cvt_f32_bf16<<<(int)((SD + 255) / 256), 256, 0, stream>>>(x,  xb,  (int)SD);
    cvt_f32_bf16<<<(int)((DD + 255) / 256), 256, 0, stream>>>(Wq, wqb, (int)DD);
    cvt_f32_bf16<<<(int)((DD + 255) / 256), 256, 0, stream>>>(Wk, wkb, (int)DD);
    cvt_f32_bf16<<<(int)((DD + 255) / 256), 256, 0, stream>>>(Wv, wvb, (int)DD);
    cvt_f32_bf16<<<(int)((DD + 255) / 256), 256, 0, stream>>>(Wo, wob, (int)DD);

    dim3 ggrid(S_LEN / TM, DIM_ / TN);

    // 2) Q = rmsnorm(x @ Wq^T + bq) + rope
    gemm_bf16<<<ggrid, 256, 0, stream>>>(xb, wqb, bq, tmp, S_LEN, DIM_, DIM_);
    norm_rope<<<S_LEN, 256, 0, stream>>>(tmp, nqw, fc, fs, qb, 1, 1);
    // 3) K = rmsnorm(x @ Wk^T + bk) + rope
    gemm_bf16<<<ggrid, 256, 0, stream>>>(xb, wkb, bk, tmp, S_LEN, DIM_, DIM_);
    norm_rope<<<S_LEN, 256, 0, stream>>>(tmp, nkw, fc, fs, kb, 1, 1);
    // 4) V = x @ Wv^T + bv   (convert only)
    gemm_bf16<<<ggrid, 256, 0, stream>>>(xb, wvb, bv, tmp, S_LEN, DIM_, DIM_);
    norm_rope<<<S_LEN, 256, 0, stream>>>(tmp, nqw, fc, fs, vb, 0, 0);

    // 5) attention
    flash_attn<<<dim3(S_LEN / 128, HEADS_), 256, 0, stream>>>(qb, kb, vb, ab, sl);

    // 6) out = attn @ Wo^T + bo  (fp32 output)
    gemm_bf16<<<ggrid, 256, 0, stream>>>(ab, wob, bo, (float*)d_out, S_LEN, DIM_, DIM_);
}
